// GlycanAA_55645596287228
// MI455X (gfx1250) — compile-verified, matrix operand order checked by reading, not run on hardware
//
#include <hip/hip_runtime.h>
#include <hip/hip_bf16.h>

#define NN 80000
#define EE 500000
#define GG 2000
#define DD 256
#define RR 4
#define LL 3
#define UNITS_C 150
#define KU 1024   // R*D

typedef __attribute__((ext_vector_type(16))) __bf16 v16bf;
typedef __attribute__((ext_vector_type(8)))  float  v8f;

__device__ __forceinline__ __bf16 f2bf(float f) {
    return (__bf16)f;   // hardware RNE convert (v_cvt_pk_bf16_f32 when paired)
}

// ---------------------------------------------------------------- utilities
__global__ void k_zero(float* __restrict__ p, size_t n4) {
    size_t i = (size_t)blockIdx.x * blockDim.x + threadIdx.x;
    size_t stride = (size_t)gridDim.x * blockDim.x;
    float4 z{0.f, 0.f, 0.f, 0.f};
    for (; i < n4; i += stride) ((float4*)p)[i] = z;
}

// x = emb[unit_type], plus atom/mono masks
__global__ void k_embed(const int* __restrict__ ut, const float* __restrict__ emb,
                        float* __restrict__ x, unsigned char* __restrict__ am,
                        unsigned char* __restrict__ mmk) {
    int t = blockIdx.x * blockDim.x + threadIdx.x;
    if (t >= NN * 64) return;
    int n = t >> 6, c = t & 63;
    int u = ut[n];
    float4 v{0.f, 0.f, 0.f, 0.f};
    if (u >= 0) v = ((const float4*)(emb + (size_t)u * DD))[c];
    ((float4*)(x + (size_t)n * DD))[c] = v;
    if (c == 0) {
        unsigned char a = (u >= UNITS_C) ? 1 : 0;
        am[n] = a;
        mmk[n] = !a;
    }
}

// per-path degree accumulation over segments dst*R+rel
__global__ void k_deg(const unsigned char* __restrict__ am,
                      const int* __restrict__ src, const int* __restrict__ dst,
                      const int* __restrict__ rel, const float* __restrict__ ew,
                      float* __restrict__ degA, float* __restrict__ degC,
                      float* __restrict__ degM) {
    int e = blockIdx.x * blockDim.x + threadIdx.x;
    if (e >= EE) return;
    bool ai = am[src[e]] != 0, ao = am[dst[e]] != 0;
    int seg = dst[e] * RR + rel[e];
    float w = ew[e];
    if (ai && ao)        atomicAdd(&degA[seg], w);
    else if (ai && !ao)  atomicAdd(&degC[seg], w);
    else if (!ai && !ao) atomicAdd(&degM[seg], w);
}

__global__ void k_norm(const unsigned char* __restrict__ am,
                       const int* __restrict__ src, const int* __restrict__ dst,
                       const int* __restrict__ rel, const float* __restrict__ ew,
                       const float* __restrict__ degA, const float* __restrict__ degC,
                       const float* __restrict__ degM,
                       float* __restrict__ wA, float* __restrict__ wC,
                       float* __restrict__ wM) {
    int e = blockIdx.x * blockDim.x + threadIdx.x;
    if (e >= EE) return;
    bool ai = am[src[e]] != 0, ao = am[dst[e]] != 0;
    int seg = dst[e] * RR + rel[e];
    float w = ew[e];
    float ea = (ai && ao) ? w : 0.f;
    float ec = (ai && !ao) ? w : 0.f;
    float em = (!ai && !ao) ? w : 0.f;
    float da = degA[seg], dc = degC[seg], dm = degM[seg];
    wA[e] = (ea > 0.f) ? ea / (da > 0.f ? da : 1.f) : 0.f;
    wC[e] = (ec > 0.f) ? ec / (dc > 0.f ? dc : 1.f) : 0.f;
    wM[e] = (em > 0.f) ? em / (dm > 0.f ? dm : 1.f) : 0.f;
}

// scatter: update[(dst*R+rel)*D + d] += x[src*D+d] * w ; 32 lanes per edge
__global__ void k_scatter(const float* __restrict__ xin, const float* __restrict__ wn,
                          const int* __restrict__ src, const int* __restrict__ dst,
                          const int* __restrict__ rel, float* __restrict__ upd) {
    int e = blockIdx.x * 8 + (threadIdx.x >> 5);
    if (e >= EE) return;
    int lane = threadIdx.x & 31;
    float w = wn[e];
    if (w == 0.f) return;
    size_t srow = (size_t)src[e] * DD;
    size_t urow = ((size_t)dst[e] * RR + rel[e]) * DD;
#pragma unroll
    for (int j = 0; j < 8; ++j) {
        int d = j * 32 + lane;
        atomicAdd(&upd[urow + d], xin[srow + d] * w);
    }
}

// pre-swizzle fp32 weights (shape [256][K], n-major) into WMMA B-fragment order:
// dst[((kt*16+nt)*32+lane)*16 + e] = bf16(W[nt*16 + lane%16][kt*32 + kmap(e,lane/16)])
__global__ void k_swzB(const float* __restrict__ W, __bf16* __restrict__ dst, int K) {
    int i = blockIdx.x * blockDim.x + threadIdx.x;
    if (i >= 256 * K) return;
    int e    = i & 15;
    int lane = (i >> 4) & 31;
    int ntkt = i >> 9;
    int nt   = ntkt & 15;
    int kt   = ntkt >> 4;
    int hlf  = lane >> 4;
    int n    = nt * 16 + (lane & 15);
    int k    = kt * 32 + (((e & 8) << 1) | (hlf * 8) | (e & 7));
    dst[i] = f2bf(W[(size_t)n * K + k]);
}

// h = relu(update @ Wu^T + b + xs @ Ws^T + bs), xs selected by mode:
//   mode 0: xs = xa ; mode 1: xs = msk ? xb : xa ; mode 2: xs = msk ? xb : 0
// Double-buffered LDS slabs of 16 rows x 128 k (4 WMMA k-steps) per barrier.
__global__ void __launch_bounds__(128)
k_gemm(const float* __restrict__ upd,
       const float* __restrict__ xa, const float* __restrict__ xb,
       const unsigned char* __restrict__ msk, int mode,
       const __bf16* __restrict__ Bu, const __bf16* __restrict__ Bs,
       const float* __restrict__ bias0, const float* __restrict__ bias1,
       float* __restrict__ hout) {
    __shared__ __align__(32) __bf16 sA[2][2048];   // 2 x (16x128) fragment-ordered
    const int t = threadIdx.x;
    const int wv = t >> 5;
    const int lane = t & 31;
    const int row0 = blockIdx.x * 16;

    v8f acc[4] = {};

    const int i   = t * 4;
    const int m   = i >> 5;    // row this thread stages
    const int kk0 = i & 31;    // first k within a 32-step it stages

    // self-term row pointer (masked select fused into staging)
    const int row = row0 + m;
    const float* rp;
    if (mode == 0)      rp = xa + (size_t)row * DD;
    else if (mode == 1) rp = (msk[row] ? xb : xa) + (size_t)row * DD;
    else                rp = msk[row] ? (xb + (size_t)row * DD) : nullptr;

    // stage one 16x128 slab of `upd` (chunk kc) into buffer `buf`
    auto stageU = [&](int buf, int kc) {
        const float* rowp = upd + (size_t)(row0 + m) * KU + kc * 128;
#pragma unroll
        for (int sub = 0; sub < 4; ++sub) {
            float4 f4 = *(const float4*)&rowp[sub * 32 + kk0];
            float fv[4] = {f4.x, f4.y, f4.z, f4.w};
#pragma unroll
            for (int j = 0; j < 4; ++j) {
                int k  = kk0 + j;
                int ln = (((k >> 3) & 1) << 4) + m;
                int e  = ((k & 16) >> 1) | (k & 7);
                sA[buf][sub * 512 + ln * 16 + e] = f2bf(fv[j]);
            }
        }
    };
    // stage one 16x128 slab of the (masked) self input
    auto stageS = [&](int buf, int kc) {
#pragma unroll
        for (int sub = 0; sub < 4; ++sub) {
            float4 f4{0.f, 0.f, 0.f, 0.f};
            if (rp) f4 = *(const float4*)&rp[kc * 128 + sub * 32 + kk0];
            float fv[4] = {f4.x, f4.y, f4.z, f4.w};
#pragma unroll
            for (int j = 0; j < 4; ++j) {
                int k  = kk0 + j;
                int ln = (((k >> 3) & 1) << 4) + m;
                int e  = ((k & 16) >> 1) | (k & 7);
                sA[buf][sub * 512 + ln * 16 + e] = f2bf(fv[j]);
            }
        }
    };
    // consume one slab: 4 k-steps x 4 n-tiles = 16 WMMAs
    auto compute = [&](int buf, const __bf16* B, int kbBase) {
#pragma unroll
        for (int ks = 0; ks < 4; ++ks) {
            v16bf afrag = ((const v16bf*)(sA[buf] + ks * 512))[lane];
#pragma unroll
            for (int q = 0; q < 4; ++q) {
                int nt = wv * 4 + q;
                v16bf bfrag = *(const v16bf*)&B[((size_t)((kbBase + ks) * 16 + nt) * 32 + lane) * 16];
                acc[q] = __builtin_amdgcn_wmma_f32_16x16x32_bf16(
                    false, afrag, false, bfrag, (short)0, acc[q], false, false);
            }
        }
    };

    // ---- update (N x 1024): 8 slabs, double buffered ----
    stageU(0, 0);
    __syncthreads();
    for (int kc = 0; kc < 8; ++kc) {
        if (kc < 7) stageU((kc + 1) & 1, kc + 1);
        else        stageS(0, 0);                 // (7+1)&1 == 0
        compute(kc & 1, Bu, kc * 4);
        __syncthreads();
    }
    // ---- self term (N x 256): 2 slabs ----
    stageS(1, 1);
    compute(0, Bs, 0);
    __syncthreads();
    compute(1, Bs, 4);

    // ---- bias + relu epilogue; C layout: lane%16 = N col, vgpr v -> M = v + 8*(lane/16)
    const int hlf = lane >> 4;
    const int nl  = lane & 15;
#pragma unroll
    for (int q = 0; q < 4; ++q) {
        int ng = (wv * 4 + q) * 16 + nl;
        float bsum = bias0[ng] + bias1[ng];
#pragma unroll
        for (int v = 0; v < 8; ++v) {
            int mr = v + hlf * 8;
            float val = acc[q][v] + bsum;
            hout[(size_t)(row0 + mr) * DD + ng] = val > 0.f ? val : 0.f;
        }
    }
}

// x = am ? h1 : h3
__global__ void k_xfinal(float* __restrict__ x, const float* __restrict__ h1,
                         const float* __restrict__ h3, const unsigned char* __restrict__ am) {
    int t = blockIdx.x * blockDim.x + threadIdx.x;
    if (t >= NN * 64) return;
    int n = t >> 6, c = t & 63;
    const float4* s = am[n] ? (const float4*)(h1 + (size_t)n * DD)
                            : (const float4*)(h3 + (size_t)n * DD);
    ((float4*)(x + (size_t)n * DD))[c] = s[c];
}

// outputs: all_node_feature = x ; node_feature = mm ? x : 0 ; graph_feature = segsum
__global__ void k_out(const float* __restrict__ x, const unsigned char* __restrict__ mmk,
                      const int* __restrict__ n2g,
                      float* __restrict__ gf, float* __restrict__ nf, float* __restrict__ anf) {
    int t = blockIdx.x * blockDim.x + threadIdx.x;
    if (t >= NN * 64) return;
    int n = t >> 6, c = t & 63;
    float4 v = ((const float4*)(x + (size_t)n * DD))[c];
    ((float4*)(anf + (size_t)n * DD))[c] = v;
    if (mmk[n]) {
        ((float4*)(nf + (size_t)n * DD))[c] = v;
        float* gp = gf + (size_t)n2g[n] * DD + c * 4;
        atomicAdd(gp + 0, v.x); atomicAdd(gp + 1, v.y);
        atomicAdd(gp + 2, v.z); atomicAdd(gp + 3, v.w);
    } else {
        float4 z{0.f, 0.f, 0.f, 0.f};
        ((float4*)(nf + (size_t)n * DD))[c] = z;
    }
}

extern "C" void kernel_launch(void* const* d_in, const int* in_sizes, int n_in,
                              void* d_out, int out_size, void* d_ws, size_t ws_size,
                              hipStream_t stream) {
    const int*   ut  = (const int*)d_in[0];
    const int*   src = (const int*)d_in[1];
    const int*   dst = (const int*)d_in[2];
    const int*   rel = (const int*)d_in[3];
    const float* ew  = (const float*)d_in[4];
    const int*   n2g = (const int*)d_in[5];
    const float* emb = (const float*)d_in[6];
    const float* Wp[3]  = {(const float*)d_in[7],  (const float*)d_in[11], (const float*)d_in[15]};
    const float* bp[3]  = {(const float*)d_in[8],  (const float*)d_in[12], (const float*)d_in[16]};
    const float* Wsp[3] = {(const float*)d_in[9],  (const float*)d_in[13], (const float*)d_in[17]};
    const float* bsp[3] = {(const float*)d_in[10], (const float*)d_in[14], (const float*)d_in[18]};

    // workspace carve-up (256B aligned)
    char* ws = (char*)d_ws;
    size_t off = 0;
    auto carve = [&](size_t bytes) -> char* {
        char* p = ws + off;
        off += (bytes + 255) & ~(size_t)255;
        return p;
    };
    float* upd = (float*)carve((size_t)NN * KU * 4);
    float* x   = (float*)carve((size_t)NN * DD * 4);
    float* h1  = (float*)carve((size_t)NN * DD * 4);
    float* h2  = (float*)carve((size_t)NN * DD * 4);
    float* h3  = (float*)carve((size_t)NN * DD * 4);
    float* deg = (float*)carve((size_t)3 * NN * RR * 4);
    float* degA = deg, *degC = deg + (size_t)NN * RR, *degM = deg + (size_t)2 * NN * RR;
    float* wn  = (float*)carve((size_t)3 * EE * 4);
    float* wA = wn, *wC = wn + EE, *wM = wn + 2 * EE;
    unsigned char* am  = (unsigned char*)carve(NN);
    unsigned char* mmk = (unsigned char*)carve(NN);
    const size_t convSz = (size_t)DD * KU + (size_t)DD * DD;   // bf16 elements per conv
    __bf16* swz = (__bf16*)carve(9 * convSz * 2);

    float* gf  = (float*)d_out;
    float* nf  = gf + (size_t)GG * DD;
    float* anf = nf + (size_t)NN * DD;

    // ---- per-call prep ----
    k_zero<<<1024, 256, 0, stream>>>(deg, (size_t)3 * NN * RR / 4);
    k_embed<<<(NN * 64 + 255) / 256, 256, 0, stream>>>(ut, emb, x, am, mmk);
    k_deg<<<(EE + 255) / 256, 256, 0, stream>>>(am, src, dst, rel, ew, degA, degC, degM);
    k_norm<<<(EE + 255) / 256, 256, 0, stream>>>(am, src, dst, rel, ew, degA, degC, degM, wA, wC, wM);

    for (int p = 0; p < 3; ++p)
        for (int l = 0; l < LL; ++l) {
            __bf16* base = swz + (size_t)(p * 3 + l) * convSz;
            k_swzB<<<KU, 256, 0, stream>>>(Wp[p]  + (size_t)l * DD * KU, base, KU);
            k_swzB<<<DD, 256, 0, stream>>>(Wsp[p] + (size_t)l * DD * DD, base + (size_t)DD * KU, DD);
        }

    const size_t updN4 = (size_t)NN * KU / 4;
    const int scGrid = (EE + 7) / 8;
    const int gmGrid = NN / 16;

    for (int l = 0; l < LL; ++l) {
        __bf16* bA = swz + (size_t)(0 * 3 + l) * convSz;
        __bf16* bC = swz + (size_t)(1 * 3 + l) * convSz;
        __bf16* bM = swz + (size_t)(2 * 3 + l) * convSz;

        // atom conv: scatter x, self-term x (mode 0) -> h1
        k_zero<<<4096, 256, 0, stream>>>(upd, updN4);
        k_scatter<<<scGrid, 256, 0, stream>>>(x, wA, src, dst, rel, upd);
        k_gemm<<<gmGrid, 128, 0, stream>>>(upd, x, nullptr, am, 0,
                                           bA, bA + (size_t)DD * KU,
                                           bp[0] + (size_t)l * DD, bsp[0] + (size_t)l * DD, h1);
        // cross conv: scatter h1 (atom srcs), self-term am?h1:x (mode 1) -> h2
        k_zero<<<4096, 256, 0, stream>>>(upd, updN4);
        k_scatter<<<scGrid, 256, 0, stream>>>(h1, wC, src, dst, rel, upd);
        k_gemm<<<gmGrid, 128, 0, stream>>>(upd, x, h1, am, 1,
                                           bC, bC + (size_t)DD * KU,
                                           bp[1] + (size_t)l * DD, bsp[1] + (size_t)l * DD, h2);
        // mono conv: scatter h2 (mono srcs), self-term mm?h2:0 (mode 2) -> h3
        k_zero<<<4096, 256, 0, stream>>>(upd, updN4);
        k_scatter<<<scGrid, 256, 0, stream>>>(h2, wM, src, dst, rel, upd);
        k_gemm<<<gmGrid, 128, 0, stream>>>(upd, nullptr, h2, mmk, 2,
                                           bM, bM + (size_t)DD * KU,
                                           bp[2] + (size_t)l * DD, bsp[2] + (size_t)l * DD, h3);
        // x = am ? h1 : h3
        k_xfinal<<<(NN * 64 + 255) / 256, 256, 0, stream>>>(x, h1, h3, am);
    }

    k_zero<<<512, 256, 0, stream>>>(gf, (size_t)GG * DD / 4);
    k_out<<<(NN * 64 + 255) / 256, 256, 0, stream>>>(x, mmk, n2g, gf, nf, anf);
}
